// MMSE_loss_79319456022842
// MI455X (gfx1250) — matrix operand order repaired and checked by reading, compiled, not run on hardware
//
#include <hip/hip_runtime.h>
#include <hip/hip_bf16.h>

#define F_BINS   257
#define T_STEPS  65536
#define N_ACT    256

typedef __attribute__((ext_vector_type(16))) _Float16 v16h;
typedef __attribute__((ext_vector_type(8)))  float    v8f;

// ---------------------------------------------------------------------------
// K0: clean_sum[t] = sum_f x_clean[f, t]   (fully coalesced column sums)
//     also zero the 256 selected-action flags.
// ---------------------------------------------------------------------------
__global__ void k0_colsum(const float* __restrict__ x_clean,
                          float* __restrict__ clean_sum,
                          int* __restrict__ sel) {
    const int t = blockIdx.x * blockDim.x + threadIdx.x;
    if (blockIdx.x == 0 && threadIdx.x < N_ACT) sel[threadIdx.x] = 0;
    float acc = 0.f;
    for (int f = 0; f < F_BINS; ++f)
        acc += x_clean[(size_t)f * T_STEPS + t];
    clean_sum[t] = acc;
}

// ---------------------------------------------------------------------------
// K1: proj[t,a] = sum_f x_source[f,t] * G[f,a]  via v_wmma_f32_16x16x32_f16,
//     fused argmin_a (clean_sum[t] - proj[t,a]) -> sel[a] = 1.
//
//     Block = 512 threads = 16 waves; each wave owns one 16-timestep tile and
//     all 256 actions (16 x v8f accumulators resident, 128 VGPRs).
//     B tiles (shared by every wave in the block) are staged in LDS once per
//     K-step in the exact per-lane WMMA layout; waves read them with
//     ds_load_b128 pairs. K = 256 via WMMA + scalar f32 tail for f = 256.
// ---------------------------------------------------------------------------
__global__ void __launch_bounds__(512)
k1_argmin(const float* __restrict__ x_source,
          const float* __restrict__ G,
          const float* __restrict__ clean_sum,
          int* __restrict__ sel) {
    // 16 KB: 16 action tiles x 32 lanes x 8 dwords (32B per lane per tile)
    __shared__ unsigned ldsB[4096];

    const int lane = threadIdx.x & 31;
    const int wave = threadIdx.x >> 5;
    const int tile = blockIdx.x * (blockDim.x >> 5) + wave;
    const int t0   = tile * 16;
    const int g    = lane >> 4;   // lane half: selects K sub-range
    const int m    = lane & 15;   // A: row M ; B: column N

    v8f C[16];
#pragma unroll
    for (int at = 0; at < 16; ++at) C[at] = (v8f){};

    for (int ks = 0; ks < 8; ++ks) {          // K-steps of 32 (f = 0..255)
        const int f0 = ks * 32;

        __syncthreads();
        // ---- cooperative stage of B slab (32 K-rows x 256 actions) -------
        // LDS word [at*256 + l*8 + w] holds f16 pair (K=2w+16*(l>>4), +1)
        // for column N = l&15 of action tile 'at'.
#pragma unroll
        for (int i = 0; i < 8; ++i) {
            const int idx = (int)threadIdx.x + i * 512;   // 0..4095
            const int kkp = idx >> 8;                     // K-row pair 0..15
            const int ac  = idx & 255;                    // action 0..255
            const int f   = f0 + kkp * 2;
            const float vlo = G[(size_t)f * N_ACT + ac];
            const float vhi = G[(size_t)(f + 1) * N_ACT + ac];
            union { _Float16 h[2]; unsigned u; } pk;
            pk.h[0] = (_Float16)vlo;
            pk.h[1] = (_Float16)vhi;
            const int l  = ((kkp >= 8) ? 16 : 0) + (ac & 15);
            const int at = ac >> 4;
            ldsB[at * 256 + l * 8 + (kkp & 7)] = pk.u;
        }
        __syncthreads();

        // ---- A tile for this K-step (16t x 32f, f16) ---------------------
        // 16-bit A layout (matches B staging): VGPR j: K = g*16+2j,(+1)
        v16h av;
#pragma unroll
        for (int j = 0; j < 8; ++j) {
            const int Klo = g * 16 + 2 * j;
            const float vlo = x_source[(size_t)(f0 + Klo)     * T_STEPS + t0 + m];
            const float vhi = x_source[(size_t)(f0 + Klo + 1) * T_STEPS + t0 + m];
            av[2 * j]     = (_Float16)vlo;
            av[2 * j + 1] = (_Float16)vhi;
        }

        // ---- 16 WMMAs: same A, B tiles streamed from LDS -----------------
#pragma unroll
        for (int at = 0; at < 16; ++at) {
            const v16h b = *(const v16h*)&ldsB[at * 256 + lane * 8];
            C[at] = __builtin_amdgcn_wmma_f32_16x16x32_f16(false, av, false, b,
                                                           (short)0, C[at],
                                                           false, false);
        }
    }

    // ---- scalar f32 tail: f = 256 (rank-1 update) ------------------------
    float xs_tail[8];
#pragma unroll
    for (int r = 0; r < 8; ++r)
        xs_tail[r] = x_source[(size_t)(F_BINS - 1) * T_STEPS + t0 + r + 8 * g];
#pragma unroll
    for (int at = 0; at < 16; ++at) {
        const float gt = G[(size_t)(F_BINS - 1) * N_ACT + at * 16 + m];
#pragma unroll
        for (int r = 0; r < 8; ++r) C[at][r] += xs_tail[r] * gt;
    }

    // ---- argmin_a (clean_sum[t] - proj[t,a]) -----------------------------
    // C layout: VGPR r, lane -> (M = r + 8*g, N = m); action = at*16 + m
    float csum[8];
#pragma unroll
    for (int r = 0; r < 8; ++r) csum[r] = clean_sum[t0 + r + 8 * g];

    float minv[8];
    int   mina[8];
#pragma unroll
    for (int r = 0; r < 8; ++r) { minv[r] = __builtin_inff(); mina[r] = 0; }

#pragma unroll
    for (int at = 0; at < 16; ++at) {
        const int a_lane = at * 16 + m;
#pragma unroll
        for (int r = 0; r < 8; ++r) {
            const float v = csum[r] - C[at][r];
            if (v < minv[r]) { minv[r] = v; mina[r] = a_lane; }  // a increasing
        }
    }

    // argmin across the 16 lanes of each half-wave (value, then lowest index)
#pragma unroll
    for (int r = 0; r < 8; ++r) {
#pragma unroll
        for (int off = 1; off < 16; off <<= 1) {
            const float ov = __shfl_xor(minv[r], off, 16);
            const int   oa = __shfl_xor(mina[r], off, 16);
            if (ov < minv[r] || (ov == minv[r] && oa < mina[r])) {
                minv[r] = ov; mina[r] = oa;
            }
        }
    }
    if (m == 0) {                    // lanes 0 and 16 cover all 16 timesteps
#pragma unroll
        for (int r = 0; r < 8; ++r) sel[mina[r]] = 1;   // identical writes
    }
}

// ---------------------------------------------------------------------------
// K2: per-column squared-error sums, block-reduced to partials (deterministic)
// ---------------------------------------------------------------------------
__global__ void k2_loss(const float* __restrict__ x_out,
                        const float* __restrict__ x_source,
                        const float* __restrict__ G,
                        const int* __restrict__ sel,
                        float* __restrict__ partials) {
    __shared__ float red[256];
    const int t = blockIdx.x * blockDim.x + threadIdx.x;
    float acc = 0.f;
    const bool pick = (t < N_ACT) && (sel[t] != 0);
    if (pick) {
        for (int f = 0; f < F_BINS; ++f) {
            const float d = G[(size_t)f * N_ACT + t] * x_source[(size_t)f * T_STEPS + t]
                          - x_out[(size_t)f * T_STEPS + t];
            acc += d * d;
        }
    } else {
        for (int f = 0; f < F_BINS; ++f) {
            const float v = x_out[(size_t)f * T_STEPS + t];
            acc += v * v;
        }
    }
    red[threadIdx.x] = acc;
    __syncthreads();
    for (int s = 128; s > 0; s >>= 1) {
        if (threadIdx.x < (unsigned)s) red[threadIdx.x] += red[threadIdx.x + s];
        __syncthreads();
    }
    if (threadIdx.x == 0) partials[blockIdx.x] = red[0];
}

// ---------------------------------------------------------------------------
// K3: final fixed-order reduction of 256 partials -> mean
// ---------------------------------------------------------------------------
__global__ void k3_final(const float* __restrict__ partials,
                         float* __restrict__ out) {
    __shared__ float red[256];
    red[threadIdx.x] = partials[threadIdx.x];
    __syncthreads();
    for (int s = 128; s > 0; s >>= 1) {
        if (threadIdx.x < (unsigned)s) red[threadIdx.x] += red[threadIdx.x + s];
        __syncthreads();
    }
    if (threadIdx.x == 0)
        out[0] = red[0] * (1.0f / ((float)T_STEPS * (float)F_BINS));
}

// ---------------------------------------------------------------------------
extern "C" void kernel_launch(void* const* d_in, const int* in_sizes, int n_in,
                              void* d_out, int out_size, void* d_ws, size_t ws_size,
                              hipStream_t stream) {
    const float* x_out    = (const float*)d_in[0];
    const float* x_source = (const float*)d_in[1];
    const float* x_clean  = (const float*)d_in[2];
    const float* G        = (const float*)d_in[3];
    float* out = (float*)d_out;

    char* ws = (char*)d_ws;
    float* clean_sum = (float*)ws;                                   // 65536 f32
    int*   sel       = (int*)(ws + (size_t)T_STEPS * 4);             // 256 i32
    float* partials  = (float*)(ws + (size_t)T_STEPS * 4 + 1024);    // 256 f32

    k0_colsum<<<T_STEPS / 256, 256, 0, stream>>>(x_clean, clean_sum, sel);
    // 4096 tiles of 16 timesteps, 16 waves (512 threads) per block
    k1_argmin<<<(T_STEPS / 16) / 16, 512, 0, stream>>>(x_source, G, clean_sum, sel);
    k2_loss<<<T_STEPS / 256, 256, 0, stream>>>(x_out, x_source, G, sel, partials);
    k3_final<<<1, 256, 0, stream>>>(partials, out);
}
